// CopyGenerator_73727408603208
// MI455X (gfx1250) — compile-verified, matrix operand order checked by reading, not run on hardware
//
#include <hip/hip_runtime.h>
#include <hip/hip_bf16.h>

// ---------------------------------------------------------------------------
// CopyGenerator fused pipeline for MI455X (gfx1250, wave32, WMMA).
//
//   K0  : fp32 -> bf16 conversion of W and hidden (float4 -> 4xbf16, b128/b64)
//   K1  : logits = hidden @ W^T + b   via v_wmma_f32_16x16x32_bf16 -> d_out
//   K2  : ONLINE per-row max/exp-sum (single 262MB sweep) + sigmoid gate
//   K3  : in-place softmax * (1-copy), float4 vectorized      (d_out)
//   K4  : scatter atomicAdd of attn*copy into vocab cols      (d_out)
//   K5a : analytic per-row total = (1-g) + g*sum(attn) - out[pad] + EPS
//   K5b : in-place log-normalize, float4 vectorized           (d_out)
// ---------------------------------------------------------------------------

#define SRC_LEN 64
#define BATCH   64
#define TLEN    32
#define D_MODEL 1024
#define VOCAB   32000
#define M_TOTAL (TLEN * BATCH)   // 2048
#define EPSV    1e-10f

typedef __attribute__((ext_vector_type(16))) __bf16 v16bf;
typedef __attribute__((ext_vector_type(8)))  __bf16 v8bf;
typedef __attribute__((ext_vector_type(4)))  __bf16 v4bf;
typedef __attribute__((ext_vector_type(8)))  float  v8f;

// ------------------------------ K0: fp32 -> bf16 (vectorized) --------------
__global__ __launch_bounds__(256)
void f32_to_bf16_kernel(const float4* __restrict__ in, v4bf* __restrict__ out, int n4) {
    int i = blockIdx.x * 256 + threadIdx.x;
    if (i < n4) {
        float4 q = in[i];
        v4bf o;
        o[0] = (__bf16)q.x; o[1] = (__bf16)q.y; o[2] = (__bf16)q.z; o[3] = (__bf16)q.w;
        out[i] = o;
    }
}

// ------------------------------ K1: WMMA GEMM ------------------------------
// C[M,N] = A[M,K] * B[N,K]^T + bias[N].  A = hidden (bf16), B = W (bf16).
// Block: 256 threads = 8 waves. Block tile 32(M) x 256(N); wave tile 16x64.
// 16-bit A-frag layout (ISA 7.12.2): lanes 0-15 -> K {0..7,16..23},
// lanes 16-31 -> K {8..15,24..31}; row = lane&15. B symmetric (col = lane&15).
__global__ __launch_bounds__(256)
void gemm_bf16_wmma_kernel(const __bf16* __restrict__ A,
                           const __bf16* __restrict__ B,
                           const float*  __restrict__ bias,
                           float*        __restrict__ C,
                           int M, int N, int K) {
    const int lane  = threadIdx.x & 31;
    const int wave  = threadIdx.x >> 5;      // 0..7
    const int waveM = wave >> 2;             // 0..1
    const int waveN = wave & 3;              // 0..3
    const int m0 = blockIdx.y * 32  + waveM * 16;
    const int n0 = blockIdx.x * 256 + waveN * 64;

    const int kHalf = (lane >= 16) ? 8 : 0;
    const int rowA  = m0 + (lane & 15);

    v8f acc[4] = {};

    for (int k0 = 0; k0 < K; k0 += 32) {
        const __bf16* pa = A + (size_t)rowA * K + k0 + kHalf;
        v8bf a_lo = *(const v8bf*)(pa);
        v8bf a_hi = *(const v8bf*)(pa + 16);
        v16bf afrag = __builtin_shufflevector(a_lo, a_hi,
            0,1,2,3,4,5,6,7,8,9,10,11,12,13,14,15);
        __builtin_prefetch(pa + 32, 0, 3);   // next K-step of A stream (near-temporal)

#pragma unroll
        for (int t = 0; t < 4; ++t) {
            const int col = n0 + t * 16 + (lane & 15);
            const __bf16* pb = B + (size_t)col * K + k0 + kHalf;
            v8bf b_lo = *(const v8bf*)(pb);
            v8bf b_hi = *(const v8bf*)(pb + 16);
            v16bf bfrag = __builtin_shufflevector(b_lo, b_hi,
                0,1,2,3,4,5,6,7,8,9,10,11,12,13,14,15);
            acc[t] = __builtin_amdgcn_wmma_f32_16x16x32_bf16(
                /*neg_a=*/false, afrag, /*neg_b=*/false, bfrag,
                /*c_mod=*/(short)0, acc[t], /*reuse_a=*/false, /*reuse_b=*/false);
        }
    }

    // C/D layout: VGPR j, lanes 0-15 -> M=j, lanes 16-31 -> M=8+j; N = lane&15.
    const int nlane = lane & 15;
    const int mbase = m0 + ((lane >> 4) ? 8 : 0);
#pragma unroll
    for (int t = 0; t < 4; ++t) {
        const int col = n0 + t * 16 + nlane;
        const float bv = bias[col];
#pragma unroll
        for (int j = 0; j < 8; ++j) {
            C[(size_t)(mbase + j) * N + col] = acc[t][j] + bv;
        }
    }
}

// --------------- K2: ONLINE row max / exp-sum / copy gate ------------------
// Single sweep over the row: running (m, s) pair with rescale on new max,
// merged across threads in the LDS tree reduction.
__global__ __launch_bounds__(256)
void row_stats_kernel(const float* __restrict__ logits,
                      float* __restrict__ rowmax, float* __restrict__ expsum,
                      float* __restrict__ copyg,
                      int N, const int* __restrict__ copy_idx_p) {
    __shared__ float mred[256];
    __shared__ float sred[256];
    const int r  = blockIdx.x;
    const int ci = copy_idx_p[0];
    const float* row = logits + (size_t)r * N;
    const float4* row4 = (const float4*)row;
    const int nvec = N >> 2;                 // 8000

    float m = -3.402823e38f;
    float s = 0.f;
    for (int v4 = threadIdx.x; v4 < nvec; v4 += 256) {
        float4 q = row4[v4];
        const int cbase = v4 << 2;
        float vals[4] = { q.x, q.y, q.z, q.w };
#pragma unroll
        for (int j = 0; j < 4; ++j) {
            float v = (cbase + j == ci) ? EPSV : vals[j];
            if (v > m) { s *= __expf(m - v); m = v; }
            s += __expf(v - m);
        }
    }
    mred[threadIdx.x] = m; sred[threadIdx.x] = s;
    __syncthreads();
    for (int t = 128; t > 0; t >>= 1) {
        if (threadIdx.x < t) {
            const float m1 = mred[threadIdx.x], s1 = sred[threadIdx.x];
            const float m2 = mred[threadIdx.x + t], s2 = sred[threadIdx.x + t];
            const float M = fmaxf(m1, m2);
            sred[threadIdx.x] = s1 * __expf(m1 - M) + s2 * __expf(m2 - M);
            mred[threadIdx.x] = M;
        }
        __syncthreads();
    }
    if (threadIdx.x == 0) {
        rowmax[r] = mred[0];
        expsum[r] = sred[0];
        copyg[r]  = 1.f / (1.f + __expf(-row[ci]));
    }
}

// --------------- K3: in-place softmax * (1-copy), float4 -------------------
__global__ __launch_bounds__(256)
void softmax_scale_kernel(float* __restrict__ logits,
                          const float* __restrict__ rowmax,
                          const float* __restrict__ expsum,
                          const float* __restrict__ copyg,
                          int N, const int* __restrict__ copy_idx_p) {
    const int r  = blockIdx.y;
    const int v4 = blockIdx.x * 256 + threadIdx.x;
    if (v4 >= (N >> 2)) return;
    const int ci = copy_idx_p[0];
    const float m     = rowmax[r];
    const float scale = (1.f - copyg[r]) / expsum[r];

    float4* p = (float4*)(logits + (size_t)r * N) + v4;
    float4 q = *p;
    const int cbase = v4 << 2;
    float vals[4] = { q.x, q.y, q.z, q.w };
#pragma unroll
    for (int j = 0; j < 4; ++j) {
        float v = (cbase + j == ci) ? EPSV : vals[j];
        vals[j] = scale * __expf(v - m);
    }
    q.x = vals[0]; q.y = vals[1]; q.z = vals[2]; q.w = vals[3];
    *p = q;
}

// --------------------------- K4: copy-scatter ------------------------------
__global__ __launch_bounds__(256)
void scatter_copy_kernel(float* __restrict__ out,
                         const float* __restrict__ attn,
                         const float* __restrict__ copyg,
                         const int* __restrict__ src,
                         const int* __restrict__ alignment, int N) {
    int i = blockIdx.x * 256 + threadIdx.x;     // [0, M_TOTAL*SRC_LEN)
    int r = i >> 6;                             // row
    int s = i & 63;                             // src position
    int b = r & (BATCH - 1);                    // batch = row % BATCH
    int word = src[s * BATCH + b];              // src[s, b, 0]
    int tgt  = alignment[word];
    float v  = attn[(size_t)r * SRC_LEN + s] * copyg[r];
    atomicAdd(out + (size_t)r * N + tgt, v);
}

// ------------------ K5a: analytic per-row normalizer -----------------------
__global__ __launch_bounds__(256)
void row_total_kernel(const float* __restrict__ out,
                      const float* __restrict__ attn,
                      const float* __restrict__ copyg,
                      float* __restrict__ total,
                      int N, const int* __restrict__ pad_idx_p) {
    int r = blockIdx.x * 256 + threadIdx.x;
    if (r >= M_TOTAL) return;
    const float g = copyg[r];
    const float4* a4 = (const float4*)(attn + (size_t)r * SRC_LEN);
    float sa = 0.f;
#pragma unroll
    for (int s = 0; s < SRC_LEN / 4; ++s) {
        float4 q = a4[s];
        sa += q.x + q.y + q.z + q.w;
    }
    const float before_pad = out[(size_t)r * N + pad_idx_p[0]];
    total[r] = (1.f - g) + g * sa - before_pad + EPSV;
}

// ------------------ K5b: in-place log-normalize, float4 --------------------
__global__ __launch_bounds__(256)
void finalize_log_kernel(float* __restrict__ out,
                         const float* __restrict__ total,
                         int N, const int* __restrict__ pad_idx_p) {
    const int r  = blockIdx.y;
    const int v4 = blockIdx.x * 256 + threadIdx.x;
    if (v4 >= (N >> 2)) return;
    const int pad = pad_idx_p[0];
    const float rt = 1.f / total[r];

    float4* p = (float4*)(out + (size_t)r * N) + v4;
    float4 q = *p;
    const int cbase = v4 << 2;
    float vals[4] = { q.x, q.y, q.z, q.w };
#pragma unroll
    for (int j = 0; j < 4; ++j) {
        float v = (cbase + j == pad) ? EPSV : vals[j];
        vals[j] = __logf(v * rt + EPSV);
    }
    q.x = vals[0]; q.y = vals[1]; q.z = vals[2]; q.w = vals[3];
    *p = q;
}

// ---------------------------------------------------------------------------
extern "C" void kernel_launch(void* const* d_in, const int* in_sizes, int n_in,
                              void* d_out, int out_size, void* d_ws, size_t ws_size,
                              hipStream_t stream) {
    // Inputs (setup_inputs order):
    const float* hidden     = (const float*)d_in[0];   // (2048, 1024)
    const float* attn       = (const float*)d_in[1];   // (2048, 64)
    const float* W          = (const float*)d_in[2];   // (32000, 1024)
    const float* bvec       = (const float*)d_in[3];   // (32000,)
    const int*   src        = (const int*)d_in[4];     // (64, 64, 1)
    const int*   alignment  = (const int*)d_in[5];     // (32000,)
    const int*   copy_idx_p = (const int*)d_in[6];     // scalar
    const int*   pad_idx_p  = (const int*)d_in[7];     // scalar

    float* out = (float*)d_out;                        // (2048, 32000) — also logits buffer

    // Workspace layout (256B aligned slabs)
    char* ws = (char*)d_ws;
    size_t off = 0;
    auto alloc = [&](size_t bytes) { void* p = ws + off; off = (off + bytes + 255) & ~(size_t)255; return p; };
    __bf16* W_bf16  = (__bf16*)alloc((size_t)VOCAB   * D_MODEL * sizeof(__bf16)); // 64 MB
    __bf16* H_bf16  = (__bf16*)alloc((size_t)M_TOTAL * D_MODEL * sizeof(__bf16)); //  4 MB
    float*  rowmax  = (float*)alloc(M_TOTAL * sizeof(float));
    float*  expsum  = (float*)alloc(M_TOTAL * sizeof(float));
    float*  copyg   = (float*)alloc(M_TOTAL * sizeof(float));
    float*  total   = (float*)alloc(M_TOTAL * sizeof(float));

    // K0: precision conversion (bf16 inputs, f32 accumulate in WMMA)
    {
        int nW4 = (VOCAB * D_MODEL) / 4;
        f32_to_bf16_kernel<<<(nW4 + 255) / 256, 256, 0, stream>>>(
            (const float4*)W, (v4bf*)W_bf16, nW4);
        int nH4 = (M_TOTAL * D_MODEL) / 4;
        f32_to_bf16_kernel<<<(nH4 + 255) / 256, 256, 0, stream>>>(
            (const float4*)hidden, (v4bf*)H_bf16, nH4);
    }

    // K1: logits = hidden @ W^T + b  -> d_out
    {
        dim3 grid(VOCAB / 256, M_TOTAL / 32);   // (125, 64)
        gemm_bf16_wmma_kernel<<<grid, 256, 0, stream>>>(H_bf16, W_bf16, bvec, out,
                                                        M_TOTAL, VOCAB, D_MODEL);
    }

    // K2: per-row stats (online, single sweep) + copy gate
    row_stats_kernel<<<M_TOTAL, 256, 0, stream>>>(out, rowmax, expsum, copyg,
                                                  VOCAB, copy_idx_p);

    // K3: in-place softmax * (1 - copy)
    {
        dim3 grid((VOCAB / 4 + 255) / 256, M_TOTAL);   // (32, 2048)
        softmax_scale_kernel<<<grid, 256, 0, stream>>>(out, rowmax, expsum, copyg,
                                                       VOCAB, copy_idx_p);
    }

    // K4: scatter-add attn*copy into gathered vocab columns
    scatter_copy_kernel<<<(M_TOTAL * SRC_LEN) / 256, 256, 0, stream>>>(
        out, attn, copyg, src, alignment, VOCAB);

    // K5a: analytic per-row renormalizer
    row_total_kernel<<<(M_TOTAL + 255) / 256, 256, 0, stream>>>(
        out, attn, copyg, total, VOCAB, pad_idx_p);

    // K5b: in-place log-normalize (pad col forced to EPS)
    {
        dim3 grid((VOCAB / 4 + 255) / 256, M_TOTAL);   // (32, 2048)
        finalize_log_kernel<<<grid, 256, 0, stream>>>(out, total, VOCAB, pad_idx_p);
    }
}
